// BiLSTMPredictor_76922864271608
// MI455X (gfx1250) — compile-verified, matrix operand order checked by reading, not run on hardware
//
#include <hip/hip_runtime.h>

// BiLSTM (3 layers, bidirectional) for MI455X / gfx1250, wave32 + WMMA bf16.
// Pipeline per (layer, dir):
//   [gemm_x]  G[t,b,4H] = X[b,t,:] @ Wih^T + bias   (time-parallel WMMA GEMM)
//   [lstm]    persistent single-WGP kernel: 512 serial steps,
//             h in LDS (bf16), c in registers, 144 wmma/step, 2 barriers/step.
// Weights are pre-swizzled into the native 16x16x32 bf16 B-fragment lane
// layout so every fragment load is a contiguous 32B-per-lane global load.
// Layer-0 input is padded to 128 cols (zero-filled) so all GEMM staging is
// branch-free wide loads; full K panel staged once per block (1 barrier).

#define BSZ 64
#define SEQ 512
#define EMB 100
#define EMBP 128                  // padded layer-0 input width
#define HID 384
#define GDIM 1536                 // 4*HID
#define NT_TOT 96                 // GDIM / 16
#define HROW 193                  // LDS h row stride in dwords (192 + 1 pad)
#define KMAXD 385                 // max staged panel row stride in dwords (24*16+1)

typedef __attribute__((ext_vector_type(16))) __bf16 v16bf;
typedef __attribute__((ext_vector_type(8)))  float  v8f;

// k index inside a 16x32 bf16 A/B fragment for (laneHalf, slot 0..15)
__device__ __forceinline__ int kmap(int lh, int slot) {
  int v = slot >> 1, hh = slot & 1;
  int k = (v < 4) ? (v * 2 + hh) : ((v - 4) * 2 + 16 + hh);
  return k + (lh ? 8 : 0);
}
// dword (bf16-pair) index inside a k-tile row for A-fragment dword p (0..7)
__device__ __forceinline__ int umap(int p, int lh) {
  return (p < 4) ? (p + lh * 4) : (p + 4 + lh * 4);
}

// ---------------------------------------------------------------- lengths
__global__ void lengths_kernel(const int* __restrict__ words,
                               int* __restrict__ lengths) {
  int b = blockIdx.x * blockDim.x + threadIdx.x;
  if (b < BSZ) {
    int c = 0;
    for (int t = 0; t < SEQ; ++t) c += (words[b * SEQ + t] != 0);
    lengths[b] = c;
  }
}

// ------------------------------------------------- embedding (pad to 128)
__global__ void embed_kernel(const int* __restrict__ words,
                             const float* __restrict__ table,
                             float* __restrict__ X0) {
  int row = blockIdx.x;                       // b*SEQ + t
  int w = words[row];
  int k = threadIdx.x;                        // blockDim == EMBP
  X0[(size_t)row * EMBP + k] = (k < EMB) ? table[(size_t)w * EMB + k] : 0.f;
}

// --------------------------------------------------- weight -> B-fragment
// dst layout: [ntile(96)][kk(nK)][lane(32)][slot(16)] bf16, element = W[n][k]
__global__ void swizzle_kernel(const float* __restrict__ W,
                               __bf16* __restrict__ dst,
                               int Ksrc, int nK) {
  int total = NT_TOT * nK * 32 * 16;
  for (int i = blockIdx.x * blockDim.x + threadIdx.x; i < total;
       i += gridDim.x * blockDim.x) {
    int slot = i & 15;
    int lane = (i >> 4) & 31;
    int frag = i >> 9;
    int kk = frag % nK;
    int nt = frag / nK;
    int n = nt * 16 + (lane & 15);
    int k = kk * 32 + kmap(lane >> 4, slot);
    float v = (k < Ksrc) ? W[(size_t)n * Ksrc + k] : 0.f;
    dst[i] = (__bf16)v;
  }
}

// ------------------------------------------------------- input projection
// X rows are padded to ld (multiple of 32), so staging is branch-free.
// G fragment layout: [t][btile(4)][ntile(96)][lane(32)][8] fp32
__global__ void __launch_bounds__(256)
gemm_x_kernel(const float* __restrict__ X, const __bf16* __restrict__ Wfrag,
              const float* __restrict__ bias, float* __restrict__ G,
              int ld, int nK) {
  __shared__ unsigned Apanel[16 * KMAXD];     // 16 rows x (nK*32) bf16, padded
  int wave = threadIdx.x >> 5, lane = threadIdx.x & 31;
  int t = blockIdx.x >> 2, btile = blockIdx.x & 3;
  int ntile = blockIdx.y * 8 + wave;
  int m = lane & 15, lh = lane >> 4;
  const v16bf* wv = (const v16bf*)Wfrag;

  // stage entire 16 x K panel once: 16 threads per row, float2 each
  int K2 = nK * 16;                           // dwords (bf16 pairs) per row
  int stride = K2 + 1;                        // odd stride: no bank conflicts
  {
    int row = threadIdx.x >> 4;
    const float* xr = X + ((size_t)(btile * 16 + row) * SEQ + t) * ld;
    unsigned* ar = Apanel + row * stride;
    for (int cp = (threadIdx.x & 15); cp < K2; cp += 16) {
      float2 x = *(const float2*)(xr + cp * 2);
      union { __bf16 h[2]; unsigned u; } pk;
      pk.h[0] = (__bf16)x.x; pk.h[1] = (__bf16)x.y;
      ar[cp] = pk.u;
    }
  }
  __syncthreads();

  v8f acc = {0.f, 0.f, 0.f, 0.f, 0.f, 0.f, 0.f, 0.f};
  const unsigned* am = Apanel + m * stride;
  for (int kk = 0; kk < nK; ++kk) {
    union { v16bf v; unsigned u[8]; } af;
#pragma unroll
    for (int p = 0; p < 8; ++p) af.u[p] = am[kk * 16 + umap(p, lh)];

    const v16bf* bp = wv + ((size_t)ntile * nK + kk) * 32 + lane;
    if (kk + 1 < nK) __builtin_prefetch((const void*)(bp + 32), 0, 1);
    v16bf bf = *bp;
    acc = __builtin_amdgcn_wmma_f32_16x16x32_bf16(
        false, af.v, false, bf, (short)0, acc, false, false);
  }

  float bv = bias[ntile * 16 + m];
  size_t gbase = (((size_t)(t * 4 + btile) * NT_TOT + ntile) * 32 + lane) * 8;
  v8f outv;
#pragma unroll
  for (int r = 0; r < 8; ++r) outv[r] = acc[r] + bv;
  *(v8f*)(G + gbase) = outv;
}

// -------------------------------------------------------------- recurrence
// One persistent workgroup (32 waves). wave = (mtile 0..3, hg 0..7).
// Wave owns batch rows [mtile*16,+16) x h-cols [hg*48,+48) for ALL 4 gates,
// so i/f/g/o quadruples and the cell state c stay in registers.
__global__ void __launch_bounds__(1024)
lstm_kernel(const float* __restrict__ G, const __bf16* __restrict__ WhhFrag,
            const int* __restrict__ lengths, float* __restrict__ Y, int dir) {
  __shared__ unsigned hbuf[BSZ * HROW];       // h, bf16, row stride 193 dwords
  __bf16* hb16 = (__bf16*)hbuf;
  const v16bf* wv = (const v16bf*)WhhFrag;

  int wave = threadIdx.x >> 5, lane = threadIdx.x & 31;
  int mtile = wave & 3, hg = wave >> 2;
  int m = lane & 15, lh = lane >> 4;

  for (int i = threadIdx.x; i < BSZ * HROW; i += 1024) hbuf[i] = 0;

  float creg[3][8];
#pragma unroll
  for (int s = 0; s < 3; ++s)
#pragma unroll
    for (int r = 0; r < 8; ++r) creg[s][r] = 0.f;

  int Lv[8];
#pragma unroll
  for (int r = 0; r < 8; ++r) Lv[r] = lengths[mtile * 16 + lh * 8 + r];

  __syncthreads();

  for (int step = 0; step < SEQ; ++step) {
    int t = dir ? (SEQ - 1 - step) : step;

    // init accumulators from precomputed x-projection fragments
    v8f acc[4][3];
#pragma unroll
    for (int g = 0; g < 4; ++g)
#pragma unroll
      for (int s = 0; s < 3; ++s) {
        int nt = g * 24 + hg * 3 + s;
        size_t gb = (((size_t)(t * 4 + mtile) * NT_TOT + nt) * 32 + lane) * 8;
        acc[g][s] = *(const v8f*)(G + gb);
      }

    // h @ Whh^T : K = 384 -> 12 k-tiles, 12 wmma each
    for (int kk = 0; kk < 12; ++kk) {
      union { v16bf v; unsigned u[8]; } af;
      int hrow = (mtile * 16 + m) * HROW + kk * 16;
#pragma unroll
      for (int p = 0; p < 8; ++p) af.u[p] = hbuf[hrow + umap(p, lh)];
#pragma unroll
      for (int g = 0; g < 4; ++g)
#pragma unroll
        for (int s = 0; s < 3; ++s) {
          int nt = g * 24 + hg * 3 + s;
          v16bf bf = wv[((size_t)nt * 12 + kk) * 32 + lane];
          acc[g][s] = __builtin_amdgcn_wmma_f32_16x16x32_bf16(
              false, af.v, false, bf, (short)0, acc[g][s], false, false);
        }
    }
    __syncthreads();   // all h reads done before h is overwritten

    // gates + cell update, fully in-register; one writer per (b, hcol)
#pragma unroll
    for (int s = 0; s < 3; ++s) {
      int hcol = hg * 48 + s * 16 + m;
#pragma unroll
      for (int r = 0; r < 8; ++r) {
        int b = mtile * 16 + lh * 8 + r;
        float iv = acc[0][s][r], fv = acc[1][s][r];
        float gv = acc[2][s][r], ov = acc[3][s][r];
        float si = 1.f / (1.f + __expf(-iv));
        float sf = 1.f / (1.f + __expf(-fv));
        float so = 1.f / (1.f + __expf(-ov));
        float tg = tanhf(gv);
        float cn = sf * creg[s][r] + si * tg;
        float hn = so * tanhf(cn);
        bool act = (t < Lv[r]);
        creg[s][r] = act ? cn : creg[s][r];
        Y[((size_t)b * SEQ + t) * (2 * HID) + hcol] = act ? hn : 0.f;
        if (act) hb16[b * (HROW * 2) + hcol] = (__bf16)hn;
      }
    }
    __syncthreads();   // h writes visible before next step's reads
  }
}

// ------------------------------------------------------------------- host
extern "C" void kernel_launch(void* const* d_in, const int* in_sizes, int n_in,
                              void* d_out, int out_size, void* d_ws,
                              size_t ws_size, hipStream_t stream) {
  (void)in_sizes; (void)n_in; (void)out_size; (void)ws_size;
  const int* words = (const int*)d_in[0];
  const float* table = (const float*)d_in[1];

  char* ws = (char*)d_ws;
  size_t off = 0;
  auto carve = [&](size_t bytes) -> char* {
    char* p = ws + off;
    off = (off + bytes + 255) & ~(size_t)255;
    return p;
  };

  int* lengths = (int*)carve(BSZ * sizeof(int));
  float* X0 = (float*)carve((size_t)BSZ * SEQ * EMBP * sizeof(float));
  float* Xa = (float*)carve((size_t)BSZ * SEQ * 2 * HID * sizeof(float));
  float* Xb = (float*)carve((size_t)BSZ * SEQ * 2 * HID * sizeof(float));
  float* G  = (float*)carve((size_t)SEQ * BSZ * GDIM * sizeof(float));

  const int ldin[3] = {EMBP, 2 * HID, 2 * HID};   // padded row strides
  const int kin[3]  = {EMB, 2 * HID, 2 * HID};    // true K for swizzle
  const int nKih[3] = {4, 24, 24};                // K tiles (32 each)
  __bf16* wih[3][2];
  __bf16* whh[3][2];
  for (int l = 0; l < 3; ++l)
    for (int d = 0; d < 2; ++d) {
      wih[l][d] = (__bf16*)carve((size_t)NT_TOT * nKih[l] * 512 * sizeof(__bf16));
      whh[l][d] = (__bf16*)carve((size_t)NT_TOT * 12 * 512 * sizeof(__bf16));
    }

  lengths_kernel<<<1, 64, 0, stream>>>(words, lengths);
  embed_kernel<<<BSZ * SEQ, EMBP, 0, stream>>>(words, table, X0);

  for (int l = 0; l < 3; ++l)
    for (int d = 0; d < 2; ++d) {
      const float* Wih = (const float*)d_in[2 + (l * 2 + d) * 3 + 0];
      const float* Whh = (const float*)d_in[2 + (l * 2 + d) * 3 + 1];
      swizzle_kernel<<<1024, 256, 0, stream>>>(Wih, wih[l][d], kin[l], nKih[l]);
      swizzle_kernel<<<1024, 256, 0, stream>>>(Whh, whh[l][d], HID, 12);
    }

  float* Xin = X0;
  for (int l = 0; l < 3; ++l) {
    float* Xout = (l == 2) ? (float*)d_out : ((l == 0) ? Xa : Xb);
    for (int d = 0; d < 2; ++d) {
      const float* bias = (const float*)d_in[2 + (l * 2 + d) * 3 + 2];
      gemm_x_kernel<<<dim3(SEQ * 4, 12), 256, 0, stream>>>(
          Xin, wih[l][d], bias, G, ldin[l], nKih[l]);
      lstm_kernel<<<1, 1024, 0, stream>>>(G, whh[l][d], lengths,
                                          Xout + d * HID, d);
    }
    Xin = Xout;
  }
}
